// SparseTransformer_27255862460458
// MI455X (gfx1250) — matrix operand order repaired and checked by reading, not compiled, not run on hardware
//
#include <hip/hip_runtime.h>

// ---------------- problem constants ----------------
#define BB   16
#define SS   512
#define DD   1024
#define HH   16
#define LL   4
#define DFF  4096
#define DKK  64
#define TOPK 51
#define NEGV (-1000000000.0f)
#define EPSV 1e-5f
#define SCALEQ 0.125f         // 1/sqrt(64)
#define MROWS (BB*SS)         // 8192

// ---------------- WMMA / TDM types ----------------
typedef __attribute__((ext_vector_type(16))) __bf16 v16bf;
typedef __attribute__((ext_vector_type(8)))  float  v8f;
typedef __attribute__((ext_vector_type(4)))  unsigned v4u;
typedef __attribute__((ext_vector_type(8)))  int    v8i;
typedef __attribute__((ext_vector_type(4)))  int    v4i;

union Frag16 {
    v16bf v;
    uint4 u4[2];
    __bf16 e[16];
};

static __device__ __forceinline__ v8f wmma_bf16(const Frag16& a, const Frag16& b, v8f c) {
    return __builtin_amdgcn_wmma_f32_16x16x32_bf16(false, a.v, false, b.v,
                                                   (short)0, c, false, false);
}
static __device__ __forceinline__ v8f v8f_zero() {
    v8f z = {0.f,0.f,0.f,0.f,0.f,0.f,0.f,0.f};
    return z;
}
static __device__ __forceinline__ unsigned lds_off(const void* p) {
    // addrspacecast(LDS->flat) keeps the LDS byte offset in the low 32 bits
    return (unsigned)(unsigned long long)p;
}

// ---- TDM: 2-D tile load (tile_rows x 64B), contiguous in LDS ----
// data_size = 8B units; row stride / tensor_dim0 given in 8B units.
static __device__ __forceinline__ void tdm_load_2d(unsigned lds_addr, const void* gaddr,
                                                   unsigned stride_units, unsigned tile_rows)
{
    unsigned long long ga = (unsigned long long)gaddr;
    v4u g0;
    g0[0] = 1u;                                   // count=1, no gather
    g0[1] = lds_addr;                             // LDS byte address
    g0[2] = (unsigned)ga;                         // global addr [31:0]
    g0[3] = (unsigned)(ga >> 32) | (2u << 30);    // addr [56:32] | type=2

    const unsigned dim1 = 1u << 20;               // generous OOB extent
    v8i g1;
    g1[0] = (int)(3u << 16);                                      // data_size=8B
    g1[1] = (int)((stride_units & 0xFFFFu) << 16);                // tensor_dim0 lo16
    g1[2] = (int)((stride_units >> 16) | ((dim1 & 0xFFFFu) << 16));// dim0 hi | dim1 lo
    g1[3] = (int)((dim1 >> 16) | (8u << 16));                     // dim1 hi | tile_dim0=8
    g1[4] = (int)(tile_rows & 0xFFFFu);                           // tile_dim1 ; tile_dim2=0
    g1[5] = (int)stride_units;                                    // tensor_dim0_stride lo32
    g1[6] = 0;
    g1[7] = 0;

    v4i z4 = {0,0,0,0};
    v8i z8 = {0,0,0,0,0,0,0,0};
    __builtin_amdgcn_tensor_load_to_lds(g0, g1, z4, z4, z8, 0);
}

// ---- per-lane async global->LDS 16B copy ----
static __device__ __forceinline__ void async_copy_b128(void* lds_dst, const void* gsrc)
{
    asm volatile("global_load_async_to_lds_b128 %0, %1, off"
                 :: "v"(lds_off(lds_dst)), "v"((unsigned long long)gsrc) : "memory");
}
static __device__ __forceinline__ void wait_async0()
{
    asm volatile("s_wait_asynccnt 0x0" ::: "memory");
}

// =====================================================================
// K0: weight transpose + f32->bf16:  Wt[N][K] = bf16(W[K][N])
// =====================================================================
__global__ __launch_bounds__(256)
void wtrans_kernel(const float* __restrict__ W, __bf16* __restrict__ Wt, int K, int N)
{
    __shared__ float tile[32][33];
    const int n0 = blockIdx.x * 32;
    const int k0 = blockIdx.y * 32;
    const int tx = threadIdx.x & 31;
    const int ty = threadIdx.x >> 5;        // 0..7
#pragma unroll
    for (int i = 0; i < 4; ++i)
        tile[ty + i * 8][tx] = W[(size_t)(k0 + ty + i * 8) * N + n0 + tx];
    __syncthreads();
#pragma unroll
    for (int i = 0; i < 4; ++i)
        Wt[(size_t)(n0 + ty + i * 8) * K + k0 + tx] = (__bf16)tile[tx][ty + i * 8];
}

// =====================================================================
// K1: h = LayerNorm(x + pos_enc), bf16 mirror, mask bias
// =====================================================================
__global__ __launch_bounds__(256)
void embed_ln_kernel(const float* __restrict__ x, const float* __restrict__ pe,
                     const float* __restrict__ g, const float* __restrict__ b,
                     float* __restrict__ hf, __bf16* __restrict__ hb,
                     float* __restrict__ mbias)
{
    const int row = blockIdx.x;
    const int s   = row & (SS - 1);
    const int t   = threadIdx.x;
    __shared__ float red[256];

    const float* xr = x + (size_t)row * DD;
    const float* pr = pe + (size_t)s * DD;

    float vals[4];
    float acc = 0.f;
#pragma unroll
    for (int i = 0; i < 4; ++i) {
        int c = t + i * 256;
        float v = xr[c] + pr[c];
        vals[i] = v;
        acc += v;
    }
    red[t] = acc; __syncthreads();
    for (int off = 128; off > 0; off >>= 1) {
        if (t < off) red[t] += red[t + off];
        __syncthreads();
    }
    const float mean = red[0] * (1.0f / DD);
    __syncthreads();

    float vacc = 0.f;
#pragma unroll
    for (int i = 0; i < 4; ++i) { float d = vals[i] - mean; vacc += d * d; }
    red[t] = vacc; __syncthreads();
    for (int off = 128; off > 0; off >>= 1) {
        if (t < off) red[t] += red[t + off];
        __syncthreads();
    }
    const float rstd = rsqrtf(red[0] * (1.0f / DD) + EPSV);

#pragma unroll
    for (int i = 0; i < 4; ++i) {
        int c = t + i * 256;
        float o = (vals[i] - mean) * rstd * g[c] + b[c];
        hf[(size_t)row * DD + c] = o;
        hb[(size_t)row * DD + c] = (__bf16)o;
    }
    if (t == 0) mbias[row] = (xr[0] == 0.0f) ? 0.0f : NEGV;
}

// =====================================================================
// K2: WMMA GEMM  C[M,N] = A_bf16[M,K] @ Wt_bf16[N,K]^T + bias
//     128x128x32 tile, 8 waves; double-buffered TDM staging pipelined
//     against WMMA via TENSORcnt (wait<=2 while next tile in flight)
// =====================================================================
#define BM 128
#define BN 128
#define BK 32

template <bool RELU, bool WF, bool WB, int N, int K>
__global__ __launch_bounds__(256)
void gemm_tpl_kernel(const __bf16* __restrict__ A, const __bf16* __restrict__ Bt,
                     const float* __restrict__ bias,
                     float* __restrict__ outf, __bf16* __restrict__ outb)
{
    __shared__ __bf16 As[2][BM][BK];   // 16 KB (buf, m, k)
    __shared__ __bf16 Bs[2][BN][BK];   // 16 KB (buf, n, k)

    const int t    = threadIdx.x;
    const int wave = t >> 5;
    const int lane = t & 31;
    const int half = lane >> 4;
    const int l16  = lane & 15;
    const int wm   = wave >> 1;
    const int wn   = wave & 1;

    const size_t row0 = (size_t)blockIdx.y * BM;
    const int    col0 = blockIdx.x * BN;
    constexpr unsigned strideU = (unsigned)(K >> 2);   // K*2 bytes / 8

    const __bf16* aTile = A  + row0 * (size_t)K;
    const __bf16* bTile = Bt + (size_t)col0 * K;

    v8f acc[2][4];
#pragma unroll
    for (int mt = 0; mt < 2; ++mt)
#pragma unroll
        for (int nt = 0; nt < 4; ++nt) acc[mt][nt] = v8f_zero();

    const unsigned asOff[2] = { lds_off(&As[0][0][0]), lds_off(&As[1][0][0]) };
    const unsigned bsOff[2] = { lds_off(&Bs[0][0][0]), lds_off(&Bs[1][0][0]) };

    // prologue: stage k0=0 into buffer 0
    if (t < 32) {
        tdm_load_2d(asOff[0], aTile, strideU, BM);
        tdm_load_2d(bsOff[0], bTile, strideU, BN);
    }

    int buf = 0;
    for (int k0 = 0; k0 < K; k0 += BK) {
        const bool has_next = (k0 + BK) < K;
        if (has_next && t < 32) {
            tdm_load_2d(asOff[buf ^ 1], aTile + k0 + BK, strideU, BM);
            tdm_load_2d(bsOff[buf ^ 1], bTile + k0 + BK, strideU, BN);
        }
        if (has_next) __builtin_amdgcn_s_wait_tensorcnt(2);   // current pair landed
        else          __builtin_amdgcn_s_wait_tensorcnt(0);
        __syncthreads();                                       // data-ready

        Frag16 afr[2];
#pragma unroll
        for (int mt = 0; mt < 2; ++mt) {
            int m = wm * 32 + mt * 16 + l16;
            afr[mt].u4[0] = *(const uint4*)&As[buf][m][half * 8];
            afr[mt].u4[1] = *(const uint4*)&As[buf][m][16 + half * 8];
        }
#pragma unroll
        for (int nt = 0; nt < 4; ++nt) {
            Frag16 bfr;
            int n = wn * 64 + nt * 16 + l16;
            bfr.u4[0] = *(const uint4*)&Bs[buf][n][half * 8];
            bfr.u4[1] = *(const uint4*)&Bs[buf][n][16 + half * 8];
#pragma unroll
            for (int mt = 0; mt < 2; ++mt)
                acc[mt][nt] = wmma_bf16(afr[mt], bfr, acc[mt][nt]);
        }
        __syncthreads();    // reads done before this buffer is overwritten
        buf ^= 1;
    }

#pragma unroll
    for (int mt = 0; mt < 2; ++mt) {
#pragma unroll
        for (int nt = 0; nt < 4; ++nt) {
            int n = col0 + wn * 64 + nt * 16 + l16;
            float bv = bias[n];
            size_t mbase = (row0 + wm * 32 + mt * 16 + half * 8) * (size_t)N + n;
#pragma unroll
            for (int r = 0; r < 8; ++r) {
                float v = acc[mt][nt][r] + bv;
                if (RELU) v = fmaxf(v, 0.0f);
                if (WF) outf[mbase + (size_t)r * N] = v;
                if (WB) outb[mbase + (size_t)r * N] = (__bf16)v;
            }
        }
    }
}

// =====================================================================
// K3: sparse attention per (batch, head, 32-query tile); 128 threads
//     LDS: Qs 4KB | Sc 64KB (reused as Vt) | Pb 32KB
// =====================================================================
#define ATTN_SMEM (4096 + 65536 + 32768)

__global__ __launch_bounds__(128)
void attn_kernel(const __bf16* __restrict__ Q, const __bf16* __restrict__ Km,
                 const __bf16* __restrict__ V, const float* __restrict__ mbias,
                 __bf16* __restrict__ out)
{
    extern __shared__ char smem[];
    __bf16 (*Qs)[64]  = (__bf16(*)[64])smem;
    float  (*Sc)[SS]  = (float (*)[SS])(smem + 4096);
    __bf16 (*Vt)[SS]  = (__bf16(*)[SS])(smem + 4096);
    __bf16 (*Pb)[SS]  = (__bf16(*)[SS])(smem + 4096 + 65536);

    const int t    = threadIdx.x;
    const int wave = t >> 5;
    const int lane = t & 31;
    const int half = lane >> 4;
    const int l16  = lane & 15;

    const int blk   = blockIdx.x;
    const int qb    = blk & 15;
    const int head  = (blk >> 4) & 15;
    const int batch = blk >> 8;
    const int q0    = qb * 32;
    const size_t base = (size_t)batch * SS;
    const int c0    = head * DKK;

    // ---- async stage Q tile 32x64 bf16 (2 x b128 per thread) ----
    {
        int r = t >> 2;
        int c = (t & 3) * 16;
        const __bf16* src = &Q[(base + q0 + r) * DD + c0 + c];
        async_copy_b128(&Qs[r][c],     src);
        async_copy_b128(&Qs[r][c + 8], src + 8);
        wait_async0();
    }
    __syncthreads();

    // ---- scores: wave handles 128 key columns ----
    const int nb = wave * 128;
    // per-kk fragment base pointers; nt stride is a constant 16*DD elems
    const __bf16* kbase0 = &Km[(base + nb + l16) * DD + c0 + half * 8];
    const __bf16* kbase1 = kbase0 + 32;
    const float*  mbbase = &mbias[base + nb + l16];
#pragma unroll
    for (int mt = 0; mt < 2; ++mt) {
        Frag16 afr[2];
#pragma unroll
        for (int kk = 0; kk < 2; ++kk) {
            int m = mt * 16 + l16;
            afr[kk].u4[0] = *(const uint4*)&Qs[m][kk * 32 + half * 8];
            afr[kk].u4[1] = *(const uint4*)&Qs[m][kk * 32 + 16 + half * 8];
        }
#pragma unroll
        for (int nt = 0; nt < 8; ++nt) {
            v8f acc = v8f_zero();
            {
                Frag16 bfr;
                bfr.u4[0] = *(const uint4*)(kbase0 + nt * (16 * DD));
                bfr.u4[1] = *(const uint4*)(kbase0 + nt * (16 * DD) + 16);
                acc = wmma_bf16(afr[0], bfr, acc);
            }
            {
                Frag16 bfr;
                bfr.u4[0] = *(const uint4*)(kbase1 + nt * (16 * DD));
                bfr.u4[1] = *(const uint4*)(kbase1 + nt * (16 * DD) + 16);
                acc = wmma_bf16(afr[1], bfr, acc);
            }
            float mb = mbbase[nt * 16];
            int n = nb + nt * 16 + l16;
#pragma unroll
            for (int r = 0; r < 8; ++r) {
                int m = mt * 16 + r + half * 8;
                Sc[m][n] = acc[r] * SCALEQ + mb;
            }
        }
    }
    __syncthreads();

    // ---- in-place transform to order-preserving keys (disjoint stripes) ----
    const int row = t >> 2;
    const int sub = t & 3;
    for (int i = sub; i < SS; i += 4) {
        unsigned kb = __float_as_uint(Sc[row][i]);
        kb = (kb & 0x80000000u) ? ~kb : (kb | 0x80000000u);
        Sc[row][i] = __uint_as_float(kb);
    }

    // ---- exact 51st-largest via 32-bit radix select (key space) ----
    unsigned p = 0u;
    for (int bit = 31; bit >= 0; --bit) {
        unsigned cand = p | (1u << bit);
        int cnt = 0;
        for (int i = sub; i < SS; i += 4)
            cnt += (__float_as_uint(Sc[row][i]) >= cand) ? 1 : 0;
        cnt += __shfl_xor(cnt, 1);
        cnt += __shfl_xor(cnt, 2);
        if (cnt >= TOPK) p = cand;
    }

    // ---- sparsify + softmax; compare in key space, invert kept values ----
    float mx = 0.0f;
    for (int i = sub; i < SS; i += 4) {
        unsigned kb = __float_as_uint(Sc[row][i]);
        float v = (kb & 0x80000000u) ? __uint_as_float(kb & 0x7fffffffu)
                                     : __uint_as_float(~kb);
        float vp = (kb >= p) ? v : 0.0f;
        mx = fmaxf(mx, vp);
    }
    mx = fmaxf(mx, __shfl_xor(mx, 1));
    mx = fmaxf(mx, __shfl_xor(mx, 2));
    float sum = 0.0f;
    for (int i = sub; i < SS; i += 4) {
        unsigned kb = __float_as_uint(Sc[row][i]);
        float v = (kb & 0x80000000u) ? __uint_as_float(kb & 0x7fffffffu)
                                     : __uint_as_float(~kb);
        float vp = (kb >= p) ? v : 0.0f;
        sum += __expf(vp - mx);
    }
    sum += __shfl_xor(sum, 1);
    sum += __shfl_xor(sum, 2);
    const float inv = 1.0f / sum;
    for (int i = sub; i < SS; i += 4) {
        unsigned kb = __float_as_uint(Sc[row][i]);
        float v = (kb & 0x80000000u) ? __uint_as_float(kb & 0x7fffffffu)
                                     : __uint_as_float(~kb);
        float vp = (kb >= p) ? v : 0.0f;
        Pb[row][i] = (__bf16)(__expf(vp - mx) * inv);
    }
    __syncthreads();   // Sc dead; reuse as Vt

    // ---- stage V transposed: Vt[dk][key] ----
    for (int i = 0; i < 32; ++i) {
        int idx = t + i * 128;
        int k   = idx >> 3;
        int cg  = (idx & 7) * 8;
        Frag16 ld;
        ld.u4[0] = *(const uint4*)&V[(base + k) * DD + c0 + cg];
#pragma unroll
        for (int j = 0; j < 8; ++j) Vt[cg + j][k] = ld.e[j];
    }
    __syncthreads();

    // ---- out(32x64) = Pb(32x512) @ V(512x64) ----
    v8f acc2[2] = { v8f_zero(), v8f_zero() };
    for (int ks = 0; ks < 16; ++ks) {
        int k0 = ks * 32;
        Frag16 bfr;
        int n = wave * 16 + l16;
        bfr.u4[0] = *(const uint4*)&Vt[n][k0 + half * 8];
        bfr.u4[1] = *(const uint4*)&Vt[n][k0 + 16 + half * 8];
#pragma unroll
        for (int mt = 0; mt < 2; ++mt) {
            Frag16 afr;
            int m = mt * 16 + l16;
            afr.u4[0] = *(const uint4*)&Pb[m][k0 + half * 8];
            afr.u4[1] = *(const uint4*)&Pb[m][k0 + 16 + half * 8];
            acc2[mt] = wmma_bf16(afr, bfr, acc2[mt]);
        }
    }
#pragma unroll
    for (int mt = 0; mt < 2; ++mt) {
        __bf16* obase = &out[(base + q0 + mt * 16 + half * 8) * DD + c0 + wave * 16 + l16];
#pragma unroll
        for (int r = 0; r < 8; ++r)
            obase[r * DD] = (__bf16)acc2[mt][r];
    }
}

// =====================================================================
// K4: h = LayerNorm(h + y); refresh bf16 mirror
// =====================================================================
__global__ __launch_bounds__(256)
void resid_ln_kernel(float* __restrict__ h, const float* __restrict__ y,
                     const float* __restrict__ g, const float* __restrict__ b,
                     __bf16* __restrict__ hb)
{
    const int row = blockIdx.x;
    const int t   = threadIdx.x;
    __shared__ float red[256];

    float vals[4];
    float acc = 0.f;
#pragma unroll
    for (int i = 0; i < 4; ++i) {
        int c = t + i * 256;
        float v = h[(size_t)row * DD + c] + y[(size_t)row * DD + c];
        vals[i] = v;
        acc += v;
    }
    red[t] = acc; __syncthreads();
    for (int off = 128; off > 0; off >>= 1) {
        if (t < off) red[t] += red[t + off];
        __syncthreads();
    }
    const float mean = red[0] * (1.0f / DD);
    __syncthreads();

    float vacc = 0.f;
#pragma unroll
    for (int i = 0; i < 4; ++i) { float d = vals[i] - mean; vacc += d * d; }
    red[t] = vacc; __syncthreads();
    for (int off = 128; off > 0; off >>= 1) {
        if (t < off) red[t] += red[t + off];
        __syncthreads();
    }
    const float rstd = rsqrtf(red[0] * (1.0f / DD) + EPSV);

#pragma unroll
    for (int i = 0; i < 4; ++i) {
        int c = t + i * 256;
        float o = (vals[i] - mean) * rstd * g[c] + b[c];
        h[(size_t)row * DD + c]  = o;
        hb[(size_t)row * DD + c] = (__bf16)o;
    }
}

// =====================================================================
// K5: out[b, d] = h[b, S-1, d]
// =====================================================================
__global__ __launch_bounds__(256)
void extract_kernel(const float* __restrict__ h, float* __restrict__ out)
{
    int i = blockIdx.x * 256 + threadIdx.x;
    int bb = i >> 10;
    int dd = i & (DD - 1);
    out[i] = h[((size_t)bb * SS + (SS - 1)) * DD + dd];
}

// =====================================================================
// host-side orchestration
// =====================================================================
extern "C" void kernel_launch(void* const* d_in, const int* in_sizes, int n_in,
                              void* d_out, int out_size, void* d_ws, size_t ws_size,
                              hipStream_t stream)
{
    const float* x     = (const float*)d_in[0];
    const float* pe    = (const float*)d_in[1];
    const float* ng    = (const float*)d_in[2];
    const float* nbi   = (const float*)d_in[3];
    const float* Wq    = (const float*)d_in[4];
    const float* bq    = (const float*)d_in[5];
    const float* Wk    = (const float*)d_in[6];
    const float* bk    = (const float*)d_in[7];
    const float* Wv    = (const float*)d_in[8];
    const float* bv    = (const float*)d_in[9];
    const float* Wo    = (const float*)d_in[10];
    const float* bo    = (const float*)d_in[11];
    const float* ln1g  = (const float*)d_in[12];
    const float* ln1b  = (const float*)d_in[13];
    const float* ln2g  = (const float*)d_in[14];
    const float* ln2b  = (const float*)d_in[15];
    const float* W1    = (const float*)d_in[16];
    const float* b1    = (const float*)d_in[17];
    const float* W2    = (const float*)d_in[18];
    const float* b2    = (const float*)d_in[19];

    char* ws = (char*)d_ws;
    const size_t szHF = (size_t)MROWS * DD * sizeof(float);    // 32 MB
    const size_t szHB = (size_t)MROWS * DD * sizeof(__bf16);   // 16 MB
    const size_t szFF = (size_t)MROWS * DFF * sizeof(__bf16);  // 64 MB

    float*  h_f    = (float*) (ws);
    __bf16* h_b    = (__bf16*)(ws + szHF);
    __bf16* q_b    = (__bf16*)(ws + szHF + szHB);
    __bf16* k_b    = (__bf16*)(ws + szHF + 2 * szHB);
    __bf16* v_b    = (__bf16*)(ws + szHF + 3 * szHB);
    __bf16* ao_b   = (__bf16*)(ws + szHF + 4 * szHB);
    float*  proj_f = (float*) (ws + szHF + 5 * szHB);
    __bf16* ff_b   = (__bf16*)(ws + 2 * szHF + 5 * szHB);
    float*  mbias  = (float*) (ws + 2 * szHF + 5 * szHB + szFF);
    __bf16* wT     = (__bf16*)(ws + 2 * szHF + 5 * szHB + szFF + 65536);

    embed_ln_kernel<<<MROWS, 256, 0, stream>>>(x, pe, ng, nbi, h_f, h_b, mbias);

    const dim3 gD  (DD  / BN, MROWS / BM);   // (8, 64)
    const dim3 gFF (DFF / BN, MROWS / BM);   // (32, 64)
    const dim3 tDD  (DD / 32, DD / 32);      // (32, 32)
    const dim3 tD1  (DFF / 32, DD / 32);     // W1: K=D,  N=DFF
    const dim3 tD2  (DD / 32, DFF / 32);     // W2: K=DFF,N=D

    for (int i = 0; i < LL; ++i) {
        const size_t wOff  = (size_t)i * DD * DD;
        const size_t w1Off = (size_t)i * DD * DFF;

        wtrans_kernel<<<tDD, 256, 0, stream>>>(Wq + wOff, wT, DD, DD);
        gemm_tpl_kernel<false, false, true, DD, DD><<<gD, 256, 0, stream>>>(
            h_b, wT, bq + i * DD, nullptr, q_b);

        wtrans_kernel<<<tDD, 256, 0, stream>>>(Wk + wOff, wT, DD, DD);
        gemm_tpl_kernel<false, false, true, DD, DD><<<gD, 256, 0, stream>>>(
            h_b, wT, bk + i * DD, nullptr, k_b);

        wtrans_kernel<<<tDD, 256, 0, stream>>>(Wv + wOff, wT, DD, DD);
        gemm_tpl_kernel<false, false, true, DD, DD><<<gD, 256, 0, stream>>>(
            h_b, wT, bv + i * DD, nullptr, v_b);

        attn_kernel<<<BB * HH * (SS / 32), 128, ATTN_SMEM, stream>>>(
            q_b, k_b, v_b, mbias, ao_b);

        wtrans_kernel<<<tDD, 256, 0, stream>>>(Wo + wOff, wT, DD, DD);
        gemm_tpl_kernel<false, true, false, DD, DD><<<gD, 256, 0, stream>>>(
            ao_b, wT, bo + i * DD, proj_f, nullptr);
        resid_ln_kernel<<<MROWS, 256, 0, stream>>>(h_f, proj_f,
                                                   ln1g + i * DD, ln1b + i * DD, h_b);

        wtrans_kernel<<<tD1, 256, 0, stream>>>(W1 + w1Off, wT, DD, DFF);
        gemm_tpl_kernel<true, false, true, DFF, DD><<<gFF, 256, 0, stream>>>(
            h_b, wT, b1 + i * DFF, nullptr, ff_b);

        wtrans_kernel<<<tD2, 256, 0, stream>>>(W2 + w1Off, wT, DFF, DD);
        gemm_tpl_kernel<false, true, false, DD, DFF><<<gD, 256, 0, stream>>>(
            ff_b, wT, b2 + i * DD, proj_f, nullptr);
        resid_ln_kernel<<<MROWS, 256, 0, stream>>>(h_f, proj_f,
                                                   ln2g + i * DD, ln2b + i * DD, h_b);
    }

    extract_kernel<<<(BB * DD) / 256, 256, 0, stream>>>(h_f, (float*)d_out);
}
